// ShiftedWindowAttention3d_42898133353493
// MI455X (gfx1250) — matrix operand order repaired and checked by reading, not compiled, not run on hardware
//
#include <hip/hip_runtime.h>

// ---------------------------------------------------------------------------
// Shifted-window 3D attention for MI455X (gfx1250, wave32, WMMA bf16).
// All GEMMs use v_wmma_f32_16x16x32_bf16; attention is flash-style (no NxN
// materialization). Relative-position bias + shift mask + padding mask are
// fused into one precomputed table (mask depends only on 3 window-class
// bits), so the softmax inner loop is load+add per element.
// Workspace usage ~134 MB.
// ---------------------------------------------------------------------------

typedef __attribute__((ext_vector_type(16))) __bf16 v16bf;
typedef __attribute__((ext_vector_type(8)))  __bf16 v8bf;
typedef __attribute__((ext_vector_type(8)))  float  v8f;

#define NW    128   // number of windows (2*8*8)
#define NH    6     // heads
#define HD    32    // head dim
#define CDIM  192
#define NTOK  392   // tokens per window (8*7*7)
#define NPAD  400   // padded rows (25 * 16)
#define KPAD  416   // padded key rows (13 * 32)

#define DEV __device__ __forceinline__

DEV unsigned short f2bf(float f) {          // f32 -> bf16, round-nearest-even
  unsigned int u = __float_as_uint(f);
  unsigned int r = ((u >> 16) & 1u) + 0x7fffu;
  return (unsigned short)((u + r) >> 16);
}

DEV v8f wmma_bf16(v16bf a, v16bf b, v8f c) {
  return __builtin_amdgcn_wmma_f32_16x16x32_bf16(false, a, false, b,
                                                 (short)0, c, false, false);
}

// A-matrix 16x32 bf16 fragment. Source row (M = lane%16) already selected by
// caller. Layout: lanes 0-15 hold K 0..15, lanes 16-31 hold K 16..31
// (matches B fragments). One contiguous 32B load.
DEV v16bf load_a_frag(const unsigned short* row_base, int lh) {
  return *(const v16bf*)(row_base + lh * 16);
}

// B-matrix 32x16 bf16 fragment. Lane gives N=lane%16 (caller applies row
// offset); K chunk = (lane/16)*16..+15, contiguous 32B in memory when the
// source is stored as src[n][k].
DEV v16bf load_b_frag(const unsigned short* n_row_base, int lh) {
  return *(const v16bf*)(n_row_base + lh * 16);
}

// 16-lane reductions (both wave halves reduce independently: xor masks < 16)
DEV float hmax16(float v) {
  v = fmaxf(v, __shfl_xor(v, 1, 32));
  v = fmaxf(v, __shfl_xor(v, 2, 32));
  v = fmaxf(v, __shfl_xor(v, 4, 32));
  v = fmaxf(v, __shfl_xor(v, 8, 32));
  return v;
}
DEV float hsum16(float v) {
  v += __shfl_xor(v, 1, 32);
  v += __shfl_xor(v, 2, 32);
  v += __shfl_xor(v, 4, 32);
  v += __shfl_xor(v, 8, 32);
  return v;
}

// ---------------------------------------------------------------------------
// Prep kernels
// ---------------------------------------------------------------------------
__global__ void prep_weights_kernel(const float* __restrict__ qkvw,
                                    const float* __restrict__ projw,
                                    unsigned short* __restrict__ Wq,
                                    unsigned short* __restrict__ Wp) {
  const int i = blockIdx.x * 256 + threadIdx.x;
  if (i < 576 * CDIM) Wq[i] = f2bf(qkvw[i]);
  if (i < CDIM * CDIM) Wp[i] = f2bf(projw[i]);
}

// Gather x with cyclic shift into per-window bf16 tiles [NW][NPAD][CDIM]
__global__ void prep_x_kernel(const float* __restrict__ x,
                              unsigned short* __restrict__ Xw) {
  const size_t i = (size_t)blockIdx.x * 256 + threadIdx.x;
  if (i >= (size_t)NW * NPAD * CDIM) return;
  const int c   = (int)(i % CDIM);
  const int row = (int)((i / CDIM) % NPAD);
  const int w   = (int)(i / ((size_t)CDIM * NPAD));
  float v = 0.f;
  if (row < NTOK) {
    const int wd = w >> 6, wh = (w >> 3) & 7, ww = w & 7;
    const int td = row / 49, rem = row - td * 49;
    const int th = rem / 7, tw = rem - th * 7;
    const int gd = (wd * 8 + td + 4) & 15;
    const int gh = (wh * 7 + th + 3) % 56;
    const int gw = (ww * 7 + tw + 3) % 56;
    v = x[(((size_t)gd * 56 + gh) * 56 + gw) * CDIM + c];
  }
  Xw[i] = f2bf(v);
}

// Fused bias table: B[head][class][m][n] = rel-pos bias + shift mask, with
// -1e30 for padded rows/cols. class bits: 4 = window at D boundary (wd==1),
// 2 = H boundary (wh==7), 1 = W boundary (ww==7). Region id inside a
// boundary window depends only on the token coordinate (< window - shift = 4 ? 1 : 2).
__global__ void prep_bias_kernel(const float* __restrict__ tbl,
                                 float* __restrict__ B) {
  const size_t i = (size_t)blockIdx.x * 256 + threadIdx.x;
  if (i >= (size_t)NH * 8 * KPAD * KPAD) return;
  const int n  = (int)(i % KPAD);
  const int m  = (int)((i / KPAD) % KPAD);
  const int hc = (int)(i / ((size_t)KPAD * KPAD));
  const int cls = hc & 7, h = hc >> 3;
  float v = -1e30f;
  if (m < NTOK && n < NTOK) {
    const int md = m / 49, mh = (m / 7) % 7, mw = m % 7;
    const int nd = n / 49, nh = (n / 7) % 7, nw = n % 7;
    const int idx = (md - nd + 7) * 169 + (mh - nh + 6) * 13 + (mw - nw + 6);
    v = tbl[idx * NH + h];
    const int rm = ((cls & 4) ? ((md < 4) ? 9 : 18) : 0) +
                   ((cls & 2) ? ((mh < 4) ? 3 : 6)  : 0) +
                   ((cls & 1) ? ((mw < 4) ? 1 : 2)  : 0);
    const int rn = ((cls & 4) ? ((nd < 4) ? 9 : 18) : 0) +
                   ((cls & 2) ? ((nh < 4) ? 3 : 6)  : 0) +
                   ((cls & 1) ? ((nw < 4) ? 1 : 2)  : 0);
    if (rm != rn) v -= 100.f;
  }
  B[i] = v;
}

// ---------------------------------------------------------------------------
// QKV GEMM: [NPAD,192] @ Wqkv^T -> Qb [w][h][KPAD][32] (pre-scaled),
// Kb [w][h][KPAD][32], Vt [w][h][32][KPAD] (transposed for PV B-frags).
// One wave per 16x64 tile; A frag reused across 4 N-tiles.
// ---------------------------------------------------------------------------
__global__ __launch_bounds__(32)
void qkv_gemm_kernel(const unsigned short* __restrict__ Xw,
                     const unsigned short* __restrict__ Wq,
                     const float* __restrict__ qkv_bias,
                     unsigned short* __restrict__ Qb,
                     unsigned short* __restrict__ Kb,
                     unsigned short* __restrict__ Vt) {
  const int w = blockIdx.x, mtile = blockIdx.y, ntb = blockIdx.z * 4;
  const int lane = threadIdx.x, lh = lane >> 4, ll = lane & 15;
  v8f acc[4];
  #pragma unroll
  for (int nt = 0; nt < 4; ++nt) acc[nt] = 0.f;

  const unsigned short* arow = Xw + ((size_t)w * NPAD + mtile * 16 + ll) * CDIM;
  #pragma unroll
  for (int kt = 0; kt < 6; ++kt) {
    const v16bf a = load_a_frag(arow + kt * 32, lh);
    #pragma unroll
    for (int nt = 0; nt < 4; ++nt) {
      const int n = (ntb + nt) * 16 + ll;
      const v16bf b = load_b_frag(Wq + (size_t)n * CDIM + kt * 32, lh);
      acc[nt] = wmma_bf16(a, b, acc[nt]);
    }
  }
  #pragma unroll
  for (int nt = 0; nt < 4; ++nt) {
    const int n = (ntb + nt) * 16 + ll;
    const float bias = qkv_bias[n];
    #pragma unroll
    for (int r = 0; r < 8; ++r) {
      const int row = mtile * 16 + r + 8 * lh;
      const float v = acc[nt][r] + bias;
      if (n < CDIM) {                                 // Q (scaled by hd^-0.5)
        const int head = n >> 5, dim = n & 31;
        Qb[(((size_t)w * NH + head) * KPAD + row) * HD + dim] =
            f2bf(v * 0.17677669529663689f);
      } else if (n < 2 * CDIM) {                      // K
        const int nn = n - CDIM, head = nn >> 5, dim = nn & 31;
        Kb[(((size_t)w * NH + head) * KPAD + row) * HD + dim] = f2bf(v);
      } else {                                        // V, transposed
        const int nn = n - 2 * CDIM, head = nn >> 5, dim = nn & 31;
        Vt[(((size_t)w * NH + head) * HD + dim) * KPAD + row] = f2bf(v);
      }
    }
  }
}

// ---------------------------------------------------------------------------
// Flash attention: one wave per (window, head, 32 query rows = 2 WMMA tiles).
// Streams 13 key tiles of 32; every K/V fragment is reused for both Q tiles.
// Online softmax with per-VGPR-slot row statistics; P transposed C->A layout
// via a 2KB LDS round trip (single-wave workgroup: barriers are S_NOP).
// ---------------------------------------------------------------------------
__global__ __launch_bounds__(32)
void attn_fa_kernel(const unsigned short* __restrict__ Qb,
                    const unsigned short* __restrict__ Kb,
                    const unsigned short* __restrict__ Vt,
                    const float* __restrict__ biasTab,
                    unsigned short* __restrict__ Ob) {
  const int pair = blockIdx.x;                 // window*NH + head
  const int w = pair / NH, head = pair % NH;
  const int mt0 = blockIdx.y * 2;              // first of two 16-row tiles
  const int lane = threadIdx.x;
  const int lh = lane >> 4, ll = lane & 15;
  const int wd = w >> 6, wh = (w >> 3) & 7, ww = w & 7;
  const int cls = ((wd == 1) ? 4 : 0) | ((wh == 7) ? 2 : 0) | ((ww == 7) ? 1 : 0);

  const unsigned short* Qbase = Qb + (size_t)(w * NH + head) * KPAD * HD;
  const unsigned short* Kbase = Kb + (size_t)(w * NH + head) * KPAD * HD;
  const unsigned short* Vbase = Vt + (size_t)(w * NH + head) * HD * KPAD;
  const float* bbase = biasTab + (size_t)(head * 8 + cls) * KPAD * KPAD;

  v16bf qa[2];
  int mrow[2][8];
  float mrun[2][8], lrun[2][8];
  v8f accA[2], accB[2];
  #pragma unroll
  for (int t = 0; t < 2; ++t) {
    const int qrow = (mt0 + t) * 16 + ll;
    qa[t] = load_a_frag(Qbase + (size_t)qrow * HD, lh);
    accA[t] = 0.f;
    accB[t] = 0.f;
    #pragma unroll
    for (int r = 0; r < 8; ++r) {
      mrow[t][r] = (mt0 + t) * 16 + r + 8 * lh;
      mrun[t][r] = -1e30f;
      lrun[t][r] = 0.f;
    }
  }

  __shared__ __attribute__((aligned(32))) unsigned short ptile[2][16 * 32];

  for (int jt = 0; jt < 13; ++jt) {
    const int kb = jt * 32;
    if (jt + 1 < 13) {
      __builtin_prefetch((const void*)(Kbase + (size_t)(kb + 32 + ll) * HD), 0, 0);
      __builtin_prefetch((const void*)(Vbase + (size_t)ll * KPAD + kb + 32), 0, 0);
    }
    // K / V fragments for this 32-key tile (shared by both Q tiles)
    const v16bf kf0 = load_b_frag(Kbase + (size_t)(kb + ll) * HD, lh);
    const v16bf kf1 = load_b_frag(Kbase + (size_t)(kb + 16 + ll) * HD, lh);
    const v16bf vb0 = load_b_frag(Vbase + (size_t)(0  + ll) * KPAD + kb, lh);
    const v16bf vb1 = load_b_frag(Vbase + (size_t)(16 + ll) * KPAD + kb, lh);
    const int n0 = kb + ll, n1 = n0 + 16;

    #pragma unroll
    for (int t = 0; t < 2; ++t) {
      const v8f z = 0.f;
      v8f s0 = wmma_bf16(qa[t], kf0, z);
      v8f s1 = wmma_bf16(qa[t], kf1, z);

      // fused bias (rel-pos + shift mask + padding mask): one load + add
      #pragma unroll
      for (int r = 0; r < 8; ++r) {
        const size_t bi = (size_t)mrow[t][r] * KPAD;
        s0[r] += bbase[bi + n0];
        s1[r] += bbase[bi + n1];
      }

      // online softmax update (row stats are per-VGPR-slot scalars)
      float corr[8];
      #pragma unroll
      for (int r = 0; r < 8; ++r) {
        const float tm = hmax16(fmaxf(s0[r], s1[r]));
        const float nm = fmaxf(mrun[t][r], tm);
        corr[r] = __expf(mrun[t][r] - nm);
        mrun[t][r] = nm;
      }
      #pragma unroll
      for (int r = 0; r < 8; ++r) {
        const float p0 = __expf(s0[r] - mrun[t][r]);
        const float p1 = __expf(s1[r] - mrun[t][r]);
        s0[r] = p0;
        s1[r] = p1;
        lrun[t][r] = lrun[t][r] * corr[r] + hsum16(p0 + p1);
        accA[t][r] *= corr[r];
        accB[t][r] *= corr[r];
      }

      // P: C-layout -> A-layout via LDS
      __syncthreads();
      #pragma unroll
      for (int r = 0; r < 8; ++r) {
        const int pr = r + 8 * lh;
        ptile[t][pr * 32 + ll]      = f2bf(s0[r]);
        ptile[t][pr * 32 + 16 + ll] = f2bf(s1[r]);
      }
      __syncthreads();
      const v16bf pa = load_a_frag(&ptile[t][ll * 32], lh);
      accA[t] = wmma_bf16(pa, vb0, accA[t]);   // O[:, 0..15]
      accB[t] = wmma_bf16(pa, vb1, accB[t]);   // O[:, 16..31]
    }
  }

  #pragma unroll
  for (int t = 0; t < 2; ++t) {
    if (mt0 + t < 25) {                        // uniform: skip padded tile 25
      #pragma unroll
      for (int r = 0; r < 8; ++r) {
        const float inv = 1.0f / lrun[t][r];
        const size_t ob = ((size_t)w * NPAD + mrow[t][r]) * CDIM + head * HD;
        Ob[ob + ll]      = f2bf(accA[t][r] * inv);
        Ob[ob + 16 + ll] = f2bf(accB[t][r] * inv);
      }
    }
  }
}

// ---------------------------------------------------------------------------
// Output projection + inverse-shift scatter to d_out (f32)
// ---------------------------------------------------------------------------
__global__ __launch_bounds__(32)
void proj_gemm_kernel(const unsigned short* __restrict__ Ob,
                      const unsigned short* __restrict__ Wp,
                      const float* __restrict__ proj_bias,
                      float* __restrict__ out) {
  const int w = blockIdx.x, mtile = blockIdx.y, ntb = blockIdx.z * 4;
  const int lane = threadIdx.x, lh = lane >> 4, ll = lane & 15;
  const int wd = w >> 6, wh = (w >> 3) & 7, ww = w & 7;
  v8f acc[4];
  #pragma unroll
  for (int nt = 0; nt < 4; ++nt) acc[nt] = 0.f;

  const unsigned short* arow = Ob + ((size_t)w * NPAD + mtile * 16 + ll) * CDIM;
  #pragma unroll
  for (int kt = 0; kt < 6; ++kt) {
    const v16bf a = load_a_frag(arow + kt * 32, lh);
    #pragma unroll
    for (int nt = 0; nt < 4; ++nt) {
      const int n = (ntb + nt) * 16 + ll;
      const v16bf b = load_b_frag(Wp + (size_t)n * CDIM + kt * 32, lh);
      acc[nt] = wmma_bf16(a, b, acc[nt]);
    }
  }
  #pragma unroll
  for (int nt = 0; nt < 4; ++nt) {
    const int n = (ntb + nt) * 16 + ll;
    const float bias = proj_bias[n];
    #pragma unroll
    for (int r = 0; r < 8; ++r) {
      const int row = mtile * 16 + r + 8 * lh;
      if (row < NTOK) {
        const int td = row / 49, rem = row - td * 49;
        const int th = rem / 7, tw = rem - th * 7;
        const int gd = (wd * 8 + td + 4) & 15;
        const int gh = (wh * 7 + th + 3) % 56;
        const int gw = (ww * 7 + tw + 3) % 56;
        out[(((size_t)gd * 56 + gh) * 56 + gw) * CDIM + n] = acc[nt][r] + bias;
      }
    }
  }
}

// ---------------------------------------------------------------------------
// Host launch
// ---------------------------------------------------------------------------
extern "C" void kernel_launch(void* const* d_in, const int* in_sizes, int n_in,
                              void* d_out, int out_size, void* d_ws, size_t ws_size,
                              hipStream_t stream) {
  const float* x     = (const float*)d_in[0];
  const float* qkvw  = (const float*)d_in[1];
  const float* qkvb  = (const float*)d_in[2];
  const float* projw = (const float*)d_in[3];
  const float* projb = (const float*)d_in[4];
  const float* rtbl  = (const float*)d_in[5];
  float* out = (float*)d_out;
  (void)in_sizes; (void)n_in; (void)out_size; (void)ws_size;

  char* ws = (char*)d_ws;
  size_t off = 0;
  auto take = [&](size_t bytes) -> void* {
    void* p = ws + off;
    off += (bytes + 255) & ~(size_t)255;
    return p;
  };
  unsigned short* Wq = (unsigned short*)take((size_t)576 * CDIM * 2);
  unsigned short* Wp = (unsigned short*)take((size_t)CDIM * CDIM * 2);
  unsigned short* Xw = (unsigned short*)take((size_t)NW * NPAD * CDIM * 2);
  unsigned short* Qb = (unsigned short*)take((size_t)NW * NH * KPAD * HD * 2);
  unsigned short* Kb = (unsigned short*)take((size_t)NW * NH * KPAD * HD * 2);
  unsigned short* Vt = (unsigned short*)take((size_t)NW * NH * HD * KPAD * 2);
  unsigned short* Ob = (unsigned short*)take((size_t)NW * NPAD * CDIM * 2);
  float* biasTab     = (float*)take((size_t)NH * 8 * KPAD * KPAD * 4);

  prep_weights_kernel<<<(576 * CDIM + 255) / 256, 256, 0, stream>>>(qkvw, projw, Wq, Wp);
  prep_x_kernel<<<(NW * NPAD * CDIM + 255) / 256, 256, 0, stream>>>(x, Xw);
  prep_bias_kernel<<<(int)(((size_t)NH * 8 * KPAD * KPAD + 255) / 256), 256, 0, stream>>>(rtbl, biasTab);
  qkv_gemm_kernel<<<dim3(NW, 25, 9), 32, 0, stream>>>(Xw, Wq, qkvb, Qb, Kb, Vt);
  attn_fa_kernel<<<dim3(NW * NH, 13), 32, 0, stream>>>(Qb, Kb, Vt, biasTab, Ob);
  proj_gemm_kernel<<<dim3(NW, 25, 3), 32, 0, stream>>>(Ob, Wp, projb, out);
}
